// DN4Layer_5841155522614
// MI455X (gfx1250) — matrix operand order repaired and checked by reading, not compiled, not run on hardware
//
#include <hip/hip_runtime.h>

#define T_ 2
#define WAY_ 10
#define SHOT_ 5
#define QUERY_ 15
#define WQ_ (WAY_*QUERY_)   /* 150 */
#define C_ 640
#define HW_ 100
#define S_ (SHOT_*HW_)      /* 500 */
#define MPAD 112            /* 7 tiles of 16 */
#define NPAD 512            /* 16 tile-pairs of 32 */
#define NEG_INF (-3.4e38f)
#define KSTEPS (C_/32)      /* 20 */

typedef __attribute__((ext_vector_type(16))) __bf16 v16bf;
typedef __attribute__((ext_vector_type(8)))  float  v8f;

#if defined(__has_builtin)
#if __has_builtin(__builtin_amdgcn_global_load_async_to_lds_b128)
#define HAVE_ASYNC_LDS 1
#endif
#endif

#ifdef HAVE_ASYNC_LDS
typedef int v4i_ __attribute__((vector_size(16)));           // matches builtin pointee
typedef __attribute__((address_space(1))) v4i_ as1_v4i;      // global
typedef __attribute__((address_space(3))) v4i_ as3_v4i;      // LDS
#define NBUF 2
#else
#define NBUF 1
#endif

// Manual flat LDS layout: Btile at offset 0 so all B chunk offsets
// (cur*40960 + kk*64 + {0,32,20480,20512} < 64K) fold into DS immediates.
#define BTILE_BYTES (NBUF * 32 * C_ * 2)
#define CSCR_OFF    BTILE_BYTES
#define CSCR_BYTES  (7 * 2 * 16 * 16 * 4)
#define PART_OFF    (CSCR_OFF + CSCR_BYTES)
#define LDS_BYTES   (PART_OFF + 16 * 4)

static __device__ __forceinline__ unsigned short f2bf(float f) {
    unsigned u = __float_as_uint(f);
    u += 0x7FFFu + ((u >> 16) & 1u);          // round to nearest even
    return (unsigned short)(u >> 16);
}

static __device__ __forceinline__ float waveRedSum(float v) {
    #pragma unroll
    for (int off = 16; off > 0; off >>= 1)
        v += __shfl_xor(v, off, 32);
    return v;
}

// branchless running top-3 insert (3 max + 2 min, no branches)
static __device__ __forceinline__ void top3ins(float v, float& t0, float& t1, float& t2) {
    float a0 = fmaxf(v, t0);
    float r0 = fminf(v, t0);
    float a1 = fmaxf(r0, t1);
    float r1 = fminf(r0, t1);
    float a2 = fmaxf(r1, t2);
    t0 = a0; t1 = a1; t2 = a2;
}

// ---------------------------------------------------------------------------
// Prep 1: query. Input q[t][wq][c][hw]; normalize over hw per (t,wq,c);
// stage A[t*WQ+wq][p=0..111][c] bf16 (rows 100..111 zero). One wave/vector.
// ---------------------------------------------------------------------------
__global__ void dn4_prep_q(const float* __restrict__ q, unsigned short* __restrict__ A) {
    int wave = blockIdx.x * (blockDim.x >> 5) + (threadIdx.x >> 5);
    int lane = threadIdx.x & 31;
    if (wave >= T_ * WQ_ * C_) return;
    int c  = wave % C_;
    int tq = wave / C_;                        // t*WQ + wq
    const float* src = q + (size_t)(tq * C_ + c) * HW_;
    float x[4];
    float ss = 0.f;
    #pragma unroll
    for (int i = 0; i < 4; ++i) {
        int p = lane + 32 * i;
        x[i] = (p < HW_) ? src[p] : 0.f;
        ss += x[i] * x[i];
    }
    ss = waveRedSum(ss);
    float inv = 1.f / fmaxf(sqrtf(ss), 1e-12f);
    unsigned short* dst = A + (size_t)tq * MPAD * C_ + c;
    #pragma unroll
    for (int i = 0; i < 4; ++i) {
        int p = lane + 32 * i;
        if (p < HW_) dst[(size_t)p * C_] = f2bf(x[i] * inv);
    }
    if (lane < MPAD - HW_) dst[(size_t)(HW_ + lane) * C_] = 0;   // zero pad rows
}

// ---------------------------------------------------------------------------
// Prep 2: support. Input s[t][w*SHOT+shot][c][hw]; normalize over c per
// (t,w,n), n = shot*hw+p; stage B[t*WAY+w][n=0..511][c] bf16 (B^T, K-contig).
// ---------------------------------------------------------------------------
__global__ void dn4_prep_s(const float* __restrict__ s, unsigned short* __restrict__ B) {
    int wave = blockIdx.x * (blockDim.x >> 5) + (threadIdx.x >> 5);
    int lane = threadIdx.x & 31;
    if (wave >= T_ * WAY_ * NPAD) return;
    int n  = wave % NPAD;
    int tw = wave / NPAD;                      // t*WAY + w
    unsigned short* dst = B + ((size_t)tw * NPAD + n) * C_;
    if (n >= S_) {                             // zero pad columns
        for (int i = 0; i < C_ / 32; ++i) dst[lane + 32 * i] = 0;
        return;
    }
    int shot = n / HW_, p = n % HW_;
    const float* src = s + ((size_t)(tw * SHOT_ + shot) * C_) * HW_ + p;
    float v[C_ / 32];
    float ss = 0.f;
    #pragma unroll
    for (int i = 0; i < C_ / 32; ++i) {
        int c = lane + 32 * i;
        v[i] = src[(size_t)c * HW_];
        ss += v[i] * v[i];
    }
    ss = waveRedSum(ss);
    float inv = 1.f / fmaxf(sqrtf(ss), 1e-12f);
    #pragma unroll
    for (int i = 0; i < C_ / 32; ++i)
        dst[lane + 32 * i] = f2bf(v[i] * inv);
}

// ---------------------------------------------------------------------------
// Main: one workgroup per (t,wq,w). 7 waves; wave i owns M-tile rows
// [16i,16i+16). A fragments resident in registers (loaded once from global);
// async double-buffered B staging in LDS; 4 independent WMMA chains; fused
// branchless top-3 using all 32 lanes. LDS ~94 KB -> multiple blocks/WGP.
// ---------------------------------------------------------------------------
__global__ void __launch_bounds__(224)
dn4_main(const unsigned short* __restrict__ A,
         const unsigned short* __restrict__ B,
         float* __restrict__ out) {
    __shared__ alignas(16) unsigned char ldsbuf[LDS_BYTES];
    unsigned short* Btile = (unsigned short*)ldsbuf;                 // [NBUF][32*C_]
    float* Cscr = (float*)(ldsbuf + CSCR_OFF);                       // [7][2][16][16]
    float* partial = (float*)(ldsbuf + PART_OFF);                    // [7]

    int bx = blockIdx.x;
    int w  = bx % WAY_;
    int tq = bx / WAY_;                        // t*WQ + wq
    int t  = tq / WQ_;
    int tid = threadIdx.x;
    int waveId = tid >> 5;
    int lane   = tid & 31;
    int halfsel = lane >> 4;
    int lm = lane & 15;

    const unsigned short* Bg0 = B + ((size_t)(t * WAY_ + w)) * NPAD * C_;
    const unsigned short* Ag0 = A + (size_t)tq * MPAD * C_;

#ifdef HAVE_ASYNC_LDS
    // async DMA: N tile-pair 0 -> buffer 0 (overlaps the global A preload below)
    for (int i = tid; i < 32 * C_ / 8; i += 224)
        __builtin_amdgcn_global_load_async_to_lds_b128(
            (as1_v4i*)(Bg0 + (size_t)i * 8),
            (as3_v4i*)(&Btile[i * 8]), 0, 0);
#endif

    // preload this wave's A fragments straight into registers (panel read once)
    int arow = waveId * 16 + lm;
    union AF { uint4 u[2]; v16bf v; };
    AF afr[KSTEPS];
    {
        const unsigned short* ag = Ag0 + arow * C_ + halfsel * 8;
        #pragma unroll
        for (int kk = 0; kk < KSTEPS; ++kk) {
            afr[kk].u[0] = *(const uint4*)(ag + kk * 32);
            afr[kk].u[1] = *(const uint4*)(ag + kk * 32 + 16);
        }
    }

    // per-lane partial top-3: lane owns row (lane&15), column-half (lane>>4)
    float t0 = NEG_INF, t1 = NEG_INF, t2 = NEG_INF;

#ifdef HAVE_ASYNC_LDS
    asm volatile("s_wait_asynccnt 0x0" ::: "memory");  // B tile 0 DMA done
    __syncthreads();
#endif

    // loop-invariant per-lane LDS bases (Btile at LDS offset 0 -> immediates)
    const unsigned short* bbase0 = &Btile[lm * C_ + halfsel * 8];
    float* crow = &Cscr[((waveId * 2) * 16 + lm) * 16 + halfsel * 8];
    float* cst  = &Cscr[((waveId * 2) * 16 + halfsel * 8) * 16 + lm];

    for (int nt = 0; nt < NPAD / 32; ++nt) {
        int cur = nt & (NBUF - 1);
#ifdef HAVE_ASYNC_LDS
        if (nt + 1 < NPAD / 32) {              // prefetch next pair into other buf
            const unsigned short* Bg = Bg0 + (size_t)(nt + 1) * 32 * C_;
            for (int i = tid; i < 32 * C_ / 8; i += 224)
                __builtin_amdgcn_global_load_async_to_lds_b128(
                    (as1_v4i*)(Bg + (size_t)i * 8),
                    (as3_v4i*)(&Btile[(NBUF - 1) * (((nt + 1) & 1) * 32 * C_) + i * 8]), 0, 0);
        }
#else
        __syncthreads();
        {
            const uint4* Bg = (const uint4*)(Bg0 + (size_t)nt * 32 * C_);
            uint4* Bs = (uint4*)Btile;
            for (int i = tid; i < 32 * C_ / 8; i += 224) Bs[i] = Bg[i];
        }
        __syncthreads();
#endif

        // 4 independent WMMA chains: {tile0,tile1} x {even-k, odd-k}
        v8f a0e = {0.f,0.f,0.f,0.f,0.f,0.f,0.f,0.f};
        v8f a0o = a0e, a1e = a0e, a1o = a0e;
        const unsigned short* bbase = bbase0 + cur * (32 * C_);
        #pragma unroll
        for (int kk = 0; kk < KSTEPS; ++kk) {
            union { uint4 u[2]; v16bf v; } b0, b1;
            const unsigned short* bp = bbase + kk * 32;
            b0.u[0] = *(const uint4*)(bp);
            b0.u[1] = *(const uint4*)(bp + 16);
            b1.u[0] = *(const uint4*)(bp + 16 * C_);
            b1.u[1] = *(const uint4*)(bp + 16 * C_ + 16);
            if (kk & 1) {
                a0o = __builtin_amdgcn_wmma_f32_16x16x32_bf16(
                    false, afr[kk].v, false, b0.v, (short)0, a0o, false, false);
                a1o = __builtin_amdgcn_wmma_f32_16x16x32_bf16(
                    false, afr[kk].v, false, b1.v, (short)0, a1o, false, false);
            } else {
                a0e = __builtin_amdgcn_wmma_f32_16x16x32_bf16(
                    false, afr[kk].v, false, b0.v, (short)0, a0e, false, false);
                a1e = __builtin_amdgcn_wmma_f32_16x16x32_bf16(
                    false, afr[kk].v, false, b1.v, (short)0, a1e, false, false);
            }
        }
        v8f acc0 = a0e + a0o;
        v8f acc1 = a1e + a1o;

        // spill both C tiles (per-wave scratch; wave-local ordering only)
        #pragma unroll
        for (int r = 0; r < 8; ++r) {
            cst[r * 16]            = acc0[r];
            cst[256 + r * 16]      = acc1[r];
        }
        asm volatile("s_wait_dscnt 0x0" ::: "memory");     // wave-local LDS fence

        // all 32 lanes scan: lane -> row (lane&15), cols halfsel*8..+7
        #pragma unroll
        for (int tile = 0; tile < 2; ++tile) {
            int nbase = (nt * 2 + tile) * 16 + halfsel * 8;
            union { uint4 u[2]; float f[8]; } cv;
            const uint4* cr = (const uint4*)(crow + tile * 256);
            cv.u[0] = cr[0];
            cv.u[1] = cr[1];
            #pragma unroll
            for (int j = 0; j < 8; ++j) {
                float v = (nbase + j < S_) ? cv.f[j] : NEG_INF;
                top3ins(v, t0, t1, t2);
            }
        }

#ifdef HAVE_ASYNC_LDS
        if (nt + 1 < NPAD / 32) {
            asm volatile("s_wait_asynccnt 0x0" ::: "memory");  // next tile ready
            __syncthreads();                                   // all waves past this buf
        }
#endif
    }

    // merge the two column-half top-3 lists for each row (lane l <-> l+16)
    float o0 = __shfl_xor(t0, 16, 32);
    float o1 = __shfl_xor(t1, 16, 32);
    float o2 = __shfl_xor(t2, 16, 32);
    top3ins(o0, t0, t1, t2);
    top3ins(o1, t0, t1, t2);
    top3ins(o2, t0, t1, t2);

    int m = waveId * 16 + lane;                // valid for lane<16
    float val = (lane < 16 && m < HW_) ? (t0 + t1 + t2) : 0.f;
    val = waveRedSum(val);
    if (lane == 0) partial[waveId] = val;
    __syncthreads();
    if (tid == 0) {
        float ssum = 0.f;
        #pragma unroll
        for (int i = 0; i < 7; ++i) ssum += partial[i];
        out[bx] = ssum;                        // out[t][wq][w] == out[bx]
    }
}

extern "C" void kernel_launch(void* const* d_in, const int* in_sizes, int n_in,
                              void* d_out, int out_size, void* d_ws, size_t ws_size,
                              hipStream_t stream) {
    (void)in_sizes; (void)n_in; (void)out_size; (void)ws_size;
    const float* q = (const float*)d_in[0];
    const float* s = (const float*)d_in[1];
    unsigned short* A = (unsigned short*)d_ws;                       // 43.0 MB
    unsigned short* B = A + (size_t)T_ * WQ_ * MPAD * C_;            // 13.1 MB

    dn4_prep_q<<<(T_ * WQ_ * C_ + 7) / 8, 256, 0, stream>>>(q, A);
    dn4_prep_s<<<(T_ * WAY_ * NPAD + 7) / 8, 256, 0, stream>>>(s, B);
    dn4_main<<<T_ * WQ_ * WAY_, 224, 0, stream>>>(A, B, (float*)d_out);
}
